// CostVolumeNetwork_58402965291353
// MI455X (gfx1250) — matrix operand order repaired and checked by reading, not compiled, not run on hardware
//
#include <hip/hip_runtime.h>

typedef __attribute__((ext_vector_type(16))) _Float16 v16h;
typedef __attribute__((ext_vector_type(4)))  _Float16 v4h;
typedef __attribute__((ext_vector_type(8)))  float    v8f;

#define MAXD   4
#define C_CH   128
#define H_DIM  128
#define W_DIM  256
#define B_DIM  8
#define TILE_W 128   // pixels per workgroup (8 waves x 16)
#define NW     8     // waves per workgroup
#define NDISP  9     // 2*MAXD+1
#define GSTRIDE 33   // padded G row stride (floats)

// Correlation cost volume via f16 WMMA (f32 accumulate).
// G[m][n] = sum_c x1[c][w0+m] * x2[c][h+dy-4][w0-4+n];  out[dy][dx][m] = G[m][m+dx]
__global__ __launch_bounds__(256)
void CostVolumeNetwork_58402965291353_kernel(const float* __restrict__ x1,
                                             const float* __restrict__ x2,
                                             float* __restrict__ out) {
  // A fragments: [kb(4)][mb(8)][lane(32)][e(16)] halves  (x1 tile, WMMA A layout)
  __shared__ __align__(32) _Float16 Afrag[4 * 8 * 32 * 16];     // 32 KB
  // B fragments: [t(9)][kb(4)][lane(32)][e(16)] halves   (x2 row window, WMMA B layout)
  __shared__ __align__(32) _Float16 Bfrag[9 * 4 * 32 * 16];     // 36 KB
  // Per-wave accumulator spill for diagonal extraction
  __shared__ float Gtile[NW * 16 * GSTRIDE];                    // ~17 KB

  const int tid  = threadIdx.x;
  const int wid  = tid >> 5;
  const int lane = tid & 31;

  const int wg    = blockIdx.x;            // b * (H*2) + h * 2 + wtile
  const int wtile = wg & 1;
  const int h     = (wg >> 1) & (H_DIM - 1);
  const int b     = wg >> 8;               // H*2 == 256
  const int w0    = wtile * TILE_W;

  const size_t chan_stride = (size_t)H_DIM * W_DIM;

  // ---------------- stage x1 tile -> A fragment layout (4 channels packed) -----------
  // A layout (16-bit 16x32): lane L: m = L&15, h2 = L>>4; elem e:
  //   e in 0..7  -> K = h2*8 + e ;  e in 8..15 -> K = 16 + h2*8 + (e-8)
  {
    const float* x1base = x1 + ((size_t)b * C_CH * H_DIM + h) * W_DIM + w0;
    #pragma unroll
    for (int i = 0; i < (32 * TILE_W) / 256; ++i) {   // 32 channel-quads * 128 cols
      int idx = tid + i * 256;
      int c   = idx & (TILE_W - 1);     // column in tile
      int kq  = idx >> 7;               // channel quad 0..31
      int k0  = kq * 4;
      const float* p = x1base + (size_t)k0 * chan_stride + c;
      float f0 = p[0];
      float f1 = p[chan_stride];
      float f2 = p[2 * chan_stride];
      float f3 = p[3 * chan_stride];
      int kb  = k0 >> 5;
      int kin = k0 & 31;
      int h2  = (kin >> 3) & 1;
      int e0  = (kin & 7) + ((kin & 16) ? 8 : 0);   // multiple of 4
      int ln  = (c & 15) + 16 * h2;
      int mb  = c >> 4;
      v4h pk = { (_Float16)f0, (_Float16)f1, (_Float16)f2, (_Float16)f3 };
      *(v4h*)&Afrag[(((kb * 8 + mb) * 32) + ln) * 16 + e0] = pk;
    }
  }
  __syncthreads();

  const float inv_c = 1.0f / (float)C_CH;
  const v16h* Av = (const v16h*)Afrag;
  const v16h* Bv = (const v16h*)Bfrag;

  for (int dy = 0; dy < NDISP; ++dy) {
    const int  hsrc  = h + dy - MAXD;
    const bool rowOK = (hsrc >= 0) && (hsrc < H_DIM);
    const float* x2base =
        x2 + ((size_t)b * C_CH * H_DIM + (rowOK ? hsrc : 0)) * W_DIM;

    // ------------- stage x2 row window (144 cols, zero padded) -> B fragments --------
    // B layout (16-bit 32x16): lane L: n = L&15, h2 = L>>4; elem e: e<8 -> K = h2*8 + e ; e>=8 -> K = 16 + h2*8 + (e-8)
    #pragma unroll
    for (int i = 0; i < (32 * 144) / 256; ++i) {      // 32 channel-quads * 144 cols
      int idx = tid + i * 256;
      int kq  = idx / 144;            // 0..31
      int j   = idx - kq * 144;       // 0..143 -> global col w0-4+j
      int k0  = kq * 4;
      int w   = w0 - MAXD + j;
      float f0 = 0.0f, f1 = 0.0f, f2 = 0.0f, f3 = 0.0f;
      if (rowOK && w >= 0 && w < W_DIM) {
        const float* p = x2base + (size_t)k0 * chan_stride + w;
        f0 = p[0];
        f1 = p[chan_stride];
        f2 = p[2 * chan_stride];
        f3 = p[3 * chan_stride];
      }
      int kb  = k0 >> 5;
      int kin = k0 & 31;
      int h2  = (kin >> 3) & 1;
      int e0  = (kin & 7) + ((kin & 16) ? 8 : 0);   // multiple of 4
      int t   = j >> 4;
      int ln  = (j & 15) + 16 * h2;
      v4h pk = { (_Float16)f0, (_Float16)f1, (_Float16)f2, (_Float16)f3 };
      *(v4h*)&Bfrag[(((t * 4 + kb) * 32) + ln) * 16 + e0] = pk;
    }
    __syncthreads();

    // ------------- WMMA: G[16 x 32] per wave, K = 128 -------------------------------
    v8f acc0 = {};
    v8f acc1 = {};
    #pragma unroll
    for (int kb = 0; kb < 4; ++kb) {
      v16h a  = Av[(kb * 8 + wid) * 32 + lane];
      v16h b0 = Bv[((wid)     * 4 + kb) * 32 + lane];
      v16h b1 = Bv[((wid + 1) * 4 + kb) * 32 + lane];
      acc0 = __builtin_amdgcn_wmma_f32_16x16x32_f16(
          false, a, false, b0, (short)0, acc0, false, false);
      acc1 = __builtin_amdgcn_wmma_f32_16x16x32_f16(
          false, a, false, b1, (short)0, acc1, false, false);
    }

    // ------------- spill accumulators (C/D layout: VGPR r -> M = r + 8*(lane>>4)) ----
    {
      float* G = &Gtile[wid * 16 * GSTRIDE];
      int mhi = (lane >> 4) * 8;
      int n   = lane & 15;
      #pragma unroll
      for (int r = 0; r < 8; ++r) {
        G[(mhi + r) * GSTRIDE + n]      = acc0[r];
        G[(mhi + r) * GSTRIDE + 16 + n] = acc1[r];
      }
    }
    __syncthreads();

    // ------------- extract diagonals: out[dy][dx][m] = G[m][m+dx] --------------------
    {
      const float* G = &Gtile[wid * 16 * GSTRIDE];
      #pragma unroll
      for (int it = 0; it < 5; ++it) {
        int idx = it * 32 + lane;          // 9 dx * 16 m = 144 outputs
        if (idx < 144) {
          int dx = idx >> 4;
          int m  = idx & 15;
          float v = G[m * GSTRIDE + (m + dx)] * inv_c;
          v = (v >= 0.0f) ? v : 0.1f * v;
          int dch = dy * NDISP + dx;
          size_t off = (((size_t)b * 81 + dch) * H_DIM + h) * W_DIM
                     + (w0 + wid * 16 + m);
          out[off] = v;
        }
      }
    }
    __syncthreads();   // all waves done with Bfrag reads + G reads before restage
  }
}

extern "C" void kernel_launch(void* const* d_in, const int* in_sizes, int n_in,
                              void* d_out, int out_size, void* d_ws, size_t ws_size,
                              hipStream_t stream) {
  (void)in_sizes; (void)n_in; (void)d_ws; (void)ws_size; (void)out_size;
  const float* x1 = (const float*)d_in[0];
  const float* x2 = (const float*)d_in[1];
  float* out = (float*)d_out;
  dim3 grid(B_DIM * H_DIM * (W_DIM / TILE_W));   // 8 * 128 * 2 = 2048 workgroups
  dim3 block(256);                               // 8 wave32s
  CostVolumeNetwork_58402965291353_kernel<<<grid, block, 0, stream>>>(x1, x2, out);
}
